// TorchLSTMModel2_2954937499773
// MI455X (gfx1250) — compile-verified, matrix operand order checked by reading, not run on hardware
//
#include <hip/hip_runtime.h>

// ---------------------------------------------------------------------------
// Fused LSTM (B=4096, T=512, I=10, H=32, O=1) for gfx1250 (MI455X, wave32).
//
// Two cooperating waves per 16-batch tile (block = 64 threads):
//   wave w owns H-columns [16w, 16w+16) and computes gate tiles {w,2+w,4+w,6+w}
//   (= i,f,g,o for its column half) with V_WMMA_F32_16X16X32_BF16, K=32=H.
//   Cell update is lane-local in the f32 D-layout. h halves are exchanged via
//   a double-buffered LDS tile; one __syncthreads per step (barrier lowering
//   waits DScnt, so write(t)->read(t) and read(t)->overwrite(t+2) are safe).
// Bias splats are hoisted out of the T loop (persistent v8f C operands), and
// x is prefetched ~8 timesteps (320 B) ahead into near caches.
// ---------------------------------------------------------------------------

typedef __attribute__((ext_vector_type(16))) __bf16 v16bf;
typedef __attribute__((ext_vector_type(8)))  float  v8f;

#define LSTM_B 4096
#define LSTM_T 512
#define LSTM_I 10
#define LSTM_H 32

__device__ __forceinline__ float fast_sigmoid(float x) {
  // 1 / (1 + 2^(-x*log2(e)))  -> v_exp_f32 + v_rcp_f32
  return __builtin_amdgcn_rcpf(1.0f + __builtin_amdgcn_exp2f(-1.442695040888963f * x));
}
__device__ __forceinline__ float fast_tanh(float x) {
  // 2 / (1 + 2^(-2x*log2(e))) - 1
  return 2.0f * __builtin_amdgcn_rcpf(1.0f + __builtin_amdgcn_exp2f(-2.885390081777926f * x)) - 1.0f;
}

__global__ __launch_bounds__(64, 1)
void lstm_wmma_kernel(const float* __restrict__ x,
                      const float* __restrict__ W_ih,
                      const float* __restrict__ W_hh,
                      const float* __restrict__ b_ih,
                      const float* __restrict__ b_hh,
                      const float* __restrict__ W_dense,
                      const float* __restrict__ b_dense,
                      float* __restrict__ out) {
  // double-buffered h exchange tile (16 rows x 32 H-cols, bf16) + output stage
  __shared__ __align__(16) __bf16 ldsH[2][16 * 32];
  __shared__ float ldsOut[2][16];

  const int lane   = threadIdx.x & 31;
  const int w      = threadIdx.x >> 5;     // cooperating wave: column half 0/1
  const int batch0 = blockIdx.x * 16;      // 16 batch rows per block
  const int col    = lane & 15;            // N (B/C/D) or M (A) index
  const int hihalf = lane >> 4;            // 0: K 0..7/16..23  1: K 8..15/24..31
  const int koff   = hihalf * 8;

  // ---- one-time: this wave's 4 gate-column weight tiles (B-layout) -------
  // gate g in {0:i,1:f,2:g,3:o} -> global tile (2g + w), columns 16*(2g+w)..+15.
  // Lane holds column n, K {koff..koff+7} in v[0..7], {koff+16..koff+23} in v[8..15].
  v16bf Wh[4], Wx[4];
  v8f   biasv[4];                              // persistent splatted C operands
#pragma unroll
  for (int g = 0; g < 4; ++g) {
    const int n = (2 * g + w) * 16 + col;
    const float* whr = W_hh + n * LSTM_H;      // row n of W_hh == col n of W_hh^T
#pragma unroll
    for (int e = 0; e < 8; ++e) {
      Wh[g][e]     = (__bf16)whr[koff + e];
      Wh[g][8 + e] = (__bf16)whr[koff + 16 + e];
    }
    const float* wxr = W_ih + n * LSTM_I;      // K padded 10 -> 32 with zeros
#pragma unroll
    for (int e = 0; e < 8; ++e) {
      const int k = koff + e;
      Wx[g][e]     = (k < LSTM_I) ? (__bf16)wxr[k] : (__bf16)0.0f;
      Wx[g][8 + e] = (__bf16)0.0f;             // K 16..31 always padding
    }
    const float b = b_ih[n] + b_hh[n];
#pragma unroll
    for (int e = 0; e < 8; ++e) biasv[g][e] = b;
  }

  // ---- recurrent state: this wave's half of c/h (f32, D-layout) ----------
  v8f cc = {0, 0, 0, 0, 0, 0, 0, 0};
  v8f hh = {0, 0, 0, 0, 0, 0, 0, 0};
  v16bf hA;                                    // full h A-tile (both halves)
#pragma unroll
  for (int e = 0; e < 16; ++e) hA[e] = (__bf16)0.0f;

  // this lane's A-row is batch row (batch0 + col); both waves stream same x
  const float* xrow = x + (size_t)(batch0 + col) * LSTM_T * LSTM_I;

  for (int t = 0; t < LSTM_T; ++t) {
    // ---- build x A-tile (16x32 bf16, K=10 valid) ----
    v16bf xA;
    const float* p = xrow + t * LSTM_I + koff;
    if (hihalf == 0) {
#pragma unroll
      for (int e = 0; e < 8; ++e) xA[e] = (__bf16)p[e];        // K 0..7
    } else {
      xA[0] = (__bf16)p[0];                                    // K 8
      xA[1] = (__bf16)p[1];                                    // K 9
#pragma unroll
      for (int e = 2; e < 8; ++e) xA[e] = (__bf16)0.0f;        // K 10..15 pad
    }
#pragma unroll
    for (int e = 8; e < 16; ++e) xA[e] = (__bf16)0.0f;         // K 16..31 pad

    if (t + 8 < LSTM_T)
      __builtin_prefetch(xrow + (t + 8) * LSTM_I, 0, 3);       // global_prefetch_b8

    // ---- gate pre-activations: 4 tiles x 2 chained WMMAs ----
    v8f acc[4];
#pragma unroll
    for (int g = 0; g < 4; ++g) {
      v8f a = __builtin_amdgcn_wmma_f32_16x16x32_bf16(
                  false, xA, false, Wx[g], (short)0, biasv[g], false, false);
      a = __builtin_amdgcn_wmma_f32_16x16x32_bf16(
                  false, hA, false, Wh[g], (short)0, a, false, false);
      acc[g] = a;
    }

    // ---- LSTM cell for this wave's column half (gate order i,f,g,o) ----
#pragma unroll
    for (int e = 0; e < 8; ++e) {
      float ig = fast_sigmoid(acc[0][e]);
      float fg = fast_sigmoid(acc[1][e]);
      float gg = fast_tanh(acc[2][e]);
      float og = fast_sigmoid(acc[3][e]);
      float c  = fg * cc[e] + ig * gg;
      cc[e]    = c;
      hh[e]    = og * fast_tanh(c);
    }

    // ---- exchange h halves: D-layout f32 -> A-layout bf16 via LDS ----
    // waves write disjoint column ranges of buffer (t&1); __syncthreads
    // (flushes DScnt before split barrier) makes write->read safe, and the
    // double buffer makes read(t) vs write(t+1) safe.
    __bf16* Lb = &ldsH[t & 1][0];
#pragma unroll
    for (int r = 0; r < 8; ++r) {
      Lb[(r + 8 * hihalf) * 32 + w * 16 + col] = (__bf16)hh[r];
    }
    __syncthreads();
    union { uint4 q[2]; v16bf v; } cv;
    cv.q[0] = *(const uint4*)(Lb + col * 32 + koff);        // K koff..koff+7
    cv.q[1] = *(const uint4*)(Lb + col * 32 + koff + 16);   // K koff+16..koff+23
    hA = cv.v;
  }

  // ---- dense head: out[b] = h_last[b,:] . W_dense[0,:] + b_dense ----
  // each wave reduces its own 16 columns (exact f32 h), then combine via LDS.
  const float wd = W_dense[w * 16 + col];
#pragma unroll
  for (int r = 0; r < 8; ++r) {
    float s = hh[r] * wd;
    s += __shfl_xor(s, 1, 32);             // butterfly over the 16-lane half
    s += __shfl_xor(s, 2, 32);
    s += __shfl_xor(s, 4, 32);
    s += __shfl_xor(s, 8, 32);
    if (col == 0) ldsOut[w][r + 8 * hihalf] = s;
  }
  __syncthreads();
  if (threadIdx.x < 16) {
    out[batch0 + threadIdx.x] =
        ldsOut[0][threadIdx.x] + ldsOut[1][threadIdx.x] + b_dense[0];
  }
}

extern "C" void kernel_launch(void* const* d_in, const int* in_sizes, int n_in,
                              void* d_out, int out_size, void* d_ws, size_t ws_size,
                              hipStream_t stream) {
  (void)in_sizes; (void)n_in; (void)out_size; (void)d_ws; (void)ws_size;
  const float* x       = (const float*)d_in[0];
  const float* W_ih    = (const float*)d_in[1];
  const float* W_hh    = (const float*)d_in[2];
  const float* b_ih    = (const float*)d_in[3];
  const float* b_hh    = (const float*)d_in[4];
  const float* W_dense = (const float*)d_in[5];
  const float* b_dense = (const float*)d_in[6];
  float* out = (float*)d_out;

  const int threads = 64;                  // 2 cooperating waves per block
  const int blocks  = LSTM_B / 16;         // 256 blocks -> 512 waves
  lstm_wmma_kernel<<<blocks, threads, 0, stream>>>(
      x, W_ih, W_hh, b_ih, b_hh, W_dense, b_dense, out);
}
